// EVL_57947698758141
// MI455X (gfx1250) — matrix operand order repaired and checked by reading, not compiled, not run on hardware
//
#include <hip/hip_runtime.h>
#include <hip/hip_bf16.h>
#include <math.h>

typedef __attribute__((ext_vector_type(16))) _Float16 v16h;
typedef __attribute__((ext_vector_type(8)))  _Float16 v8h;
typedef __attribute__((ext_vector_type(8)))  float    v8f;

#define NB 2
#define G 64
#define GP 66
#define SP_N (G*G*G)            // 262144
#define HID 128
#define NUM_CLASS 34
#define DET_THRESH 0.2f
#define CAP 4096
#define TOPK 128
#define OUTW 43                 // 9 + 34

// ---------------------------------------------------------------------------
// prep: f32 NCDHW -> padded channels-last f16  [B][66][66][66][64]
// ---------------------------------------------------------------------------
__global__ __launch_bounds__(256) void k_prep_input(const float* __restrict__ neck,
                                                    _Float16* __restrict__ xh) {
  long i = (long)blockIdx.x * 256 + threadIdx.x;     // [b][c][sp]
  if (i >= (long)NB * 64 * SP_N) return;
  int sp = (int)(i & (SP_N - 1));
  int c  = (int)((i >> 18) & 63);
  int b  = (int)(i >> 24);
  int x = sp & 63, y = (sp >> 6) & 63, z = sp >> 12;
  size_t o = ((((size_t)b * GP + (z + 1)) * GP + (y + 1)) * GP + (x + 1)) * 64 + c;
  xh[o] = (_Float16)neck[i];
}

// weights [N][CIN][27] f32 -> [N][27*CIN] f16 (tap-major K)
template <int CIN>
__global__ __launch_bounds__(256) void k_prep_wt(const float* __restrict__ w,
                                                 _Float16* __restrict__ wt) {
  int i = blockIdx.x * 256 + threadIdx.x;
  if (i >= HID * CIN * 27) return;
  int t = i % 27;
  int c = (i / 27) % CIN;
  int n = i / (27 * CIN);
  wt[(size_t)n * 27 * CIN + t * CIN + c] = (_Float16)w[i];
}

// ---------------------------------------------------------------------------
// implicit-GEMM 3x3x3 conv, WMMA f16->f32, fused bias+ReLU, f16 output
// block: 256 thr (8 waves); tile = 32 voxels (x) x 128 out channels
// ---------------------------------------------------------------------------
template <int CIN>
__global__ __launch_bounds__(256) void k_conv3x3(const _Float16* __restrict__ Xp,
                                                 const _Float16* __restrict__ Wt,
                                                 const float* __restrict__ bias,
                                                 _Float16* __restrict__ Y,
                                                 int outPadded) {
  __shared__ alignas(16) _Float16 As[32][32];
  __shared__ alignas(16) _Float16 Bs[128][32];
  const int K = 27 * CIN;
  int bx = blockIdx.x;
  int x0 = (bx & 1) * 32;
  int y  = (bx >> 1) & 63;
  int z  = (bx >> 7) & 63;
  int b  = bx >> 13;
  int tid  = threadIdx.x;
  int lane = tid & 31, wave = tid >> 5;
  int m16 = wave & 1;     // which 16-voxel M block
  int nq  = wave >> 1;    // which pair of 16-chan N blocks
  int l16 = lane & 15, hl = lane >> 4;
  v8f acc0 = {}; v8f acc1 = {};

  for (int t = 0; t < 27; ++t) {
    int dz = t / 9, dy = (t / 3) % 3, dx = t % 3;
    const _Float16* xrow =
        Xp + ((((size_t)b * GP + (z + dz)) * GP + (y + dy)) * GP + (x0 + dx)) * CIN;
    __builtin_prefetch(xrow + 34 * CIN, 0, 1);   // hint next row into L2
    for (int cc = 0; cc < CIN; cc += 32) {
      __syncthreads();
      if (tid < 128) {                     // stage A: 32 voxels x 32 k
        int m = tid >> 2, c8 = (tid & 3) * 8;
        *(v8h*)(&As[m][c8]) = *(const v8h*)(xrow + (size_t)m * CIN + cc + c8);
      }
      {                                    // stage B: 128 n x 32 k
        int n = tid >> 1, c16 = (tid & 1) * 16;
        const _Float16* wr = Wt + (size_t)n * K + t * CIN + cc + c16;
        *(v8h*)(&Bs[n][c16])     = *(const v8h*)(wr);
        *(v8h*)(&Bs[n][c16 + 8]) = *(const v8h*)(wr + 8);
      }
      __syncthreads();
      int mv = m16 * 16 + l16;
      int ao = hl ? 16 : 0;                // A frag: lanes<16 K0-15, else K16-31
      v16h a;
      ((v8h*)&a)[0] = *(const v8h*)(&As[mv][ao]);
      ((v8h*)&a)[1] = *(const v8h*)(&As[mv][ao + 8]);
      int bo  = hl ? 16 : 0;               // B frag: lanes<16 K0-15, else K16-31
      int bn0 = (nq * 2) * 16 + l16;
      int bn1 = bn0 + 16;
      v16h fb0, fb1;
      ((v8h*)&fb0)[0] = *(const v8h*)(&Bs[bn0][bo]);
      ((v8h*)&fb0)[1] = *(const v8h*)(&Bs[bn0][bo + 8]);
      ((v8h*)&fb1)[0] = *(const v8h*)(&Bs[bn1][bo]);
      ((v8h*)&fb1)[1] = *(const v8h*)(&Bs[bn1][bo + 8]);
      acc0 = __builtin_amdgcn_wmma_f32_16x16x32_f16(false, a, false, fb0, (short)0, acc0, false, false);
      acc1 = __builtin_amdgcn_wmma_f32_16x16x32_f16(false, a, false, fb1, (short)0, acc1, false, false);
    }
  }
  // epilogue: C/D layout -> lane l16 = N, vgpr v -> M = v + 8*hl
  int on0 = (nq * 2) * 16 + l16, on1 = on0 + 16;
  float bv0 = bias[on0], bv1 = bias[on1];
#pragma unroll
  for (int v = 0; v < 8; ++v) {
    int mv = m16 * 16 + v + 8 * hl;
    int xo = x0 + mv;
    size_t ob;
    if (outPadded)
      ob = ((((size_t)b * GP + (z + 1)) * GP + (y + 1)) * GP + (xo + 1)) * (size_t)HID;
    else
      ob = ((((size_t)b * G + z) * G + y) * G + xo) * (size_t)HID;
    float r0 = acc0[v] + bv0; r0 = r0 > 0.f ? r0 : 0.f;
    float r1 = acc1[v] + bv1; r1 = r1 > 0.f ? r1 : 0.f;
    Y[ob + on0] = (_Float16)r0;
    Y[ob + on1] = (_Float16)r1;
  }
}

// ---------------------------------------------------------------------------
// final 1x1 conv + per-head activation.  MODE 0: sigmoid->cent; 1: bbox xform; 2: softmax
// ---------------------------------------------------------------------------
template <int F, int MODE>
__global__ __launch_bounds__(256) void k_head_final(const _Float16* __restrict__ H2,
                                                    const float* __restrict__ wf,
                                                    const float* __restrict__ bf,
                                                    float* __restrict__ out) {
  __shared__ float wfs[F * HID];
  for (int j = threadIdx.x; j < F * HID; j += 256) wfs[j] = wf[j];
  __syncthreads();
  int i = blockIdx.x * 256 + threadIdx.x;            // [b][sp]
  if (i >= NB * SP_N) return;
  int sp = i & (SP_N - 1);
  int b  = i >> 18;
  const _Float16* xp = H2 + (size_t)i * HID;
  float acc[F];
#pragma unroll
  for (int f = 0; f < F; ++f) acc[f] = bf[f];
  for (int c = 0; c < HID; ++c) {
    float x = (float)xp[c];
#pragma unroll
    for (int f = 0; f < F; ++f) acc[f] = fmaf(x, wfs[f * HID + c], acc[f]);
  }
  if (MODE == 0) {
    out[i] = 1.0f / (1.0f + expf(-acc[0]));
  } else if (MODE == 1) {
#pragma unroll
    for (int f = 0; f < F; ++f) {
      float v = acc[f], r;
      if (f < 3)      r = 5.9f * (1.0f / (1.0f + expf(-v))) + 0.1f;   // sizes
      else if (f < 6) r = 0.2f * tanhf(v);                            // offsets
      else            r = 1.6f * tanhf(v);                            // yaw
      out[((size_t)b * F + f) * SP_N + sp] = r;
    }
  } else {
    float mx = acc[0];
#pragma unroll
    for (int f = 1; f < F; ++f) mx = fmaxf(mx, acc[f]);
    float s = 0.f;
#pragma unroll
    for (int f = 0; f < F; ++f) { acc[f] = expf(acc[f] - mx); s += acc[f]; }
    float inv = 1.0f / s;
#pragma unroll
    for (int f = 0; f < F; ++f) out[((size_t)b * F + f) * SP_N + sp] = acc[f] * inv;
  }
}

// ---------------------------------------------------------------------------
// NMS helpers
// ---------------------------------------------------------------------------
__global__ __launch_bounds__(256) void k_maxpool5(const float* __restrict__ in,
                                                  float* __restrict__ out) {
  int i = blockIdx.x * 256 + threadIdx.x;
  if (i >= NB * SP_N) return;
  int sp = i & (SP_N - 1);
  int b  = i >> 18;
  int x = sp & 63, y = (sp >> 6) & 63, z = sp >> 12;
  const float* base = in + (size_t)b * SP_N;
  float m = -INFINITY;
  for (int dz = -2; dz <= 2; ++dz) {
    int zz = z + dz; if (zz < 0 || zz > 63) continue;
    for (int dy = -2; dy <= 2; ++dy) {
      int yy = y + dy; if (yy < 0 || yy > 63) continue;
      for (int dx = -2; dx <= 2; ++dx) {
        int xx = x + dx; if (xx < 0 || xx > 63) continue;
        m = fmaxf(m, base[((zz * 64 + yy) * 64) + xx]);
      }
    }
  }
  out[i] = m;
}

__global__ void k_eq_mask(const float* a, const float* b, float* m, int n) {
  int i = blockIdx.x * 256 + threadIdx.x;
  if (i < n) m[i] = (a[i] == b[i]) ? 1.0f : 0.0f;
}
__global__ void k_suppress(const float* sc, const float* mpm, float* ss, int n) {
  int i = blockIdx.x * 256 + threadIdx.x;
  if (i < n) ss[i] = (mpm[i] > 0.0f) ? 0.0f : sc[i];
}
__global__ void k_mask_update(float* mask, const float* ss, const float* mps,
                              const float* mpm, int n) {
  int i = blockIdx.x * 256 + threadIdx.x;
  if (i < n) {
    if ((ss[i] == mps[i]) && !(mpm[i] > 0.0f)) mask[i] = 1.0f;
  }
}
__global__ void k_final_nms(const float* sc, const float* mask, float* out, int n) {
  int i = blockIdx.x * 256 + threadIdx.x;
  if (i < n) out[i] = (mask[i] > 0.0f) ? sc[i] : 0.0f;
}

// ---------------------------------------------------------------------------
// top-k: compact candidates > thresh, then per-batch selection sort
// ---------------------------------------------------------------------------
__global__ void k_collect(const float* __restrict__ nms, float* cs, int* ci, int* cnt) {
  int i = blockIdx.x * 256 + threadIdx.x;
  if (i >= NB * SP_N) return;
  float v = nms[i];
  if (v > DET_THRESH) {
    int b = i >> 18;
    int p = atomicAdd(&cnt[b], 1);
    if (p < CAP) { cs[b * CAP + p] = v; ci[b * CAP + p] = i & (SP_N - 1); }
  }
}

__global__ __launch_bounds__(256) void k_select(float* __restrict__ cs,
                                                const int* __restrict__ ci,
                                                const int* __restrict__ cnt,
                                                int* __restrict__ selI,
                                                float* __restrict__ selS) {
  __shared__ float bs[256];
  __shared__ int   bp[256];
  int b   = blockIdx.x;
  int tid = threadIdx.x;
  int n = cnt[b]; if (n > CAP) n = CAP;
  float* mycs = cs + b * CAP;
  const int* myci = ci + b * CAP;
  for (int k = 0; k < TOPK; ++k) {
    float bestS = -3e30f; int bestP = -1;
    for (int j = tid; j < n; j += 256) {
      float s = mycs[j];
      if (s > bestS || (s == bestS && bestP >= 0 && myci[j] < myci[bestP])) {
        bestS = s; bestP = j;
      }
    }
    bs[tid] = bestS; bp[tid] = bestP;
    __syncthreads();
    for (int off = 128; off > 0; off >>= 1) {
      if (tid < off) {
        float s2 = bs[tid + off]; int p2 = bp[tid + off];
        bool take = (s2 > bs[tid]) ||
                    (s2 == bs[tid] && p2 >= 0 &&
                     (bp[tid] < 0 || myci[p2] < myci[bp[tid]]));
        if (take) { bs[tid] = s2; bp[tid] = p2; }
      }
      __syncthreads();
    }
    if (tid == 0) {
      int p = bp[0];
      if (k < n && p >= 0 && bs[0] > -1e29f) {
        selI[b * TOPK + k] = myci[p];
        selS[b * TOPK + k] = bs[0];
        mycs[p] = -2e30f;            // mark picked
      } else {
        selI[b * TOPK + k] = -1;
        selS[b * TOPK + k] = 0.0f;
      }
    }
    __syncthreads();
  }
}

__global__ void k_assemble(const int* __restrict__ selI, const float* __restrict__ selS,
                           const float* __restrict__ bbox, const float* __restrict__ clas,
                           float* __restrict__ out) {
  int t = blockIdx.x * 256 + threadIdx.x;
  if (t >= NB * TOPK) return;
  int b = t >> 7, k = t & 127;
  float* row = out + (size_t)(b * TOPK + k) * OUTW;
  int idx = selI[t];
  if (idx < 0) {
    for (int j = 0; j < OUTW; ++j) row[j] = 0.0f;
    return;
  }
  int ix = idx & 63, iy = (idx >> 6) & 63, iz = idx >> 12;
  float cx = -3.2f + ((float)ix + 0.5f) * 0.1f;
  float cy = -3.2f + ((float)iy + 0.5f) * 0.1f;
  float cz = -3.2f + ((float)iz + 0.5f) * 0.1f;
  const float* bb = bbox + (size_t)b * 7 * SP_N;
  float s0 = bb[0 * SP_N + idx], s1 = bb[1 * SP_N + idx], s2 = bb[2 * SP_N + idx];
  float ox = bb[3 * SP_N + idx], oy = bb[4 * SP_N + idx], oz = bb[5 * SP_N + idx];
  float yw = bb[6 * SP_N + idx];
  row[0] = cx + ox; row[1] = cy + oy; row[2] = cz + oz;
  row[3] = s0; row[4] = s1; row[5] = s2;
  row[6] = yw; row[7] = selS[t]; row[8] = 1.0f;
  const float* cp = clas + (size_t)b * NUM_CLASS * SP_N;
  for (int f = 0; f < NUM_CLASS; ++f) row[9 + f] = cp[(size_t)f * SP_N + idx];
}

// ---------------------------------------------------------------------------
// host
// ---------------------------------------------------------------------------
static inline size_t align256(size_t v) { return (v + 255) & ~(size_t)255; }

extern "C" void kernel_launch(void* const* d_in, const int* in_sizes, int n_in,
                              void* d_out, int out_size, void* d_ws, size_t ws_size,
                              hipStream_t stream) {
  (void)in_sizes; (void)n_in; (void)out_size; (void)ws_size;
  char* w = (char*)d_ws;
  size_t off = 0;
  const size_t XH_B   = (size_t)NB * GP * GP * GP * 64  * 2;   // 73.6 MB padded f16 input
  const size_t H1P_B  = (size_t)NB * GP * GP * GP * HID * 2;   // 147 MB padded f16
  const size_t H2_B   = (size_t)NB * SP_N * HID * 2;           // 134 MB f16
  const size_t WT_B   = (size_t)HID * 27 * HID * 2;            // max K=3456
  const size_t SPF_B  = (size_t)NB * SP_N * 4;

  _Float16* XH  = (_Float16*)(w + off); off += align256(XH_B);
  _Float16* H1P = (_Float16*)(w + off); off += align256(H1P_B);
  _Float16* H2  = (_Float16*)(w + off); off += align256(H2_B);
  _Float16* WT  = (_Float16*)(w + off); off += align256(WT_B);
  float* CENT = (float*)(w + off); off += align256(SPF_B);
  float* T1   = (float*)(w + off); off += align256(SPF_B);
  float* T2   = (float*)(w + off); off += align256(SPF_B);
  float* MASK = (float*)(w + off); off += align256(SPF_B);
  float* SS   = (float*)(w + off); off += align256(SPF_B);
  float* NMSO = (float*)(w + off); off += align256(SPF_B);
  float* BBOX = (float*)(w + off); off += align256((size_t)NB * 7 * SP_N * 4);
  float* CLAS = (float*)(w + off); off += align256((size_t)NB * NUM_CLASS * SP_N * 4);
  float* CS   = (float*)(w + off); off += align256((size_t)NB * CAP * 4);
  int*   CI   = (int*)  (w + off); off += align256((size_t)NB * CAP * 4);
  int*   CNT  = (int*)  (w + off); off += align256(256);
  int*   SELI = (int*)  (w + off); off += align256((size_t)NB * TOPK * 4);
  float* SELS = (float*)(w + off); off += align256((size_t)NB * TOPK * 4);

  // halo zeroing (interior fully rewritten each call)
  hipMemsetAsync(XH, 0, XH_B, stream);
  hipMemsetAsync(H1P, 0, H1P_B, stream);
  hipMemsetAsync(CNT, 0, 256, stream);

  const int NV = NB * SP_N;                      // 524288
  k_prep_input<<<(NB * 64 * SP_N) / 256, 256, 0, stream>>>((const float*)d_in[0], XH);

  const int NCONV = NB * 64 * 64 * 2;            // 16384 blocks

  auto run_convs = [&](int base) {
    k_prep_wt<64><<<(HID * 64 * 27 + 255) / 256, 256, 0, stream>>>(
        (const float*)d_in[base], WT);
    k_conv3x3<64><<<NCONV, 256, 0, stream>>>(XH, WT, (const float*)d_in[base + 1], H1P, 1);
    k_prep_wt<128><<<(HID * 128 * 27 + 255) / 256, 256, 0, stream>>>(
        (const float*)d_in[base + 2], WT);
    k_conv3x3<128><<<NCONV, 256, 0, stream>>>(H1P, WT, (const float*)d_in[base + 3], H2, 0);
  };

  // cent head
  run_convs(1);
  k_head_final<1, 0><<<NV / 256, 256, 0, stream>>>(H2, (const float*)d_in[5],
                                                   (const float*)d_in[6], CENT);
  // bbox head
  run_convs(7);
  k_head_final<7, 1><<<NV / 256, 256, 0, stream>>>(H2, (const float*)d_in[11],
                                                   (const float*)d_in[12], BBOX);
  // clas head
  run_convs(13);
  k_head_final<34, 2><<<NV / 256, 256, 0, stream>>>(H2, (const float*)d_in[17],
                                                    (const float*)d_in[18], CLAS);

  // simple_nms3d
  int g = NV / 256;
  k_maxpool5<<<g, 256, 0, stream>>>(CENT, T1);
  k_eq_mask<<<g, 256, 0, stream>>>(CENT, T1, MASK, NV);
  for (int r = 0; r < 2; ++r) {
    k_maxpool5<<<g, 256, 0, stream>>>(MASK, T1);          // supp source
    k_suppress<<<g, 256, 0, stream>>>(CENT, T1, SS, NV);
    k_maxpool5<<<g, 256, 0, stream>>>(SS, T2);
    k_mask_update<<<g, 256, 0, stream>>>(MASK, SS, T2, T1, NV);
  }
  k_final_nms<<<g, 256, 0, stream>>>(CENT, MASK, NMSO, NV);

  // top-k + gather
  k_collect<<<g, 256, 0, stream>>>(NMSO, CS, CI, CNT);
  k_select<<<NB, 256, 0, stream>>>(CS, CI, CNT, SELI, SELS);
  k_assemble<<<1, 256, 0, stream>>>(SELI, SELS, BBOX, CLAS, (float*)d_out);
}
